// NeuralSDE_39032662786219
// MI455X (gfx1250) — matrix operand order repaired and checked by reading, not compiled, not run on hardware
//
#include <hip/hip_runtime.h>
#include <hip/hip_bf16.h>

// ---------------- problem constants ----------------
#define T_STEPS 1000
#define BATCH   256
#define IN_CH   32
#define HID     128
#define OUT_CH  10
#define BN_EPS  1e-5f

typedef __attribute__((ext_vector_type(16))) _Float16 v16h;
typedef __attribute__((ext_vector_type(8)))  _Float16 v8h;
typedef __attribute__((ext_vector_type(8)))  float    v8f;
typedef __attribute__((ext_vector_type(4)))  unsigned v4u;
typedef __attribute__((ext_vector_type(8)))  int      v8i;
typedef __attribute__((ext_vector_type(4)))  int      v4i;

#if __has_builtin(__builtin_amdgcn_tensor_load_to_lds) && __has_builtin(__builtin_amdgcn_s_wait_tensorcnt)
#define USE_TDM 1
#else
#define USE_TDM 0
#endif

// ---------------- fast tanh (v_tanh_f32 if available, else exp2-based) ----
__device__ __forceinline__ float fast_tanh(float x) {
#if __has_builtin(__builtin_amdgcn_tanhf)
    return __builtin_amdgcn_tanhf(x);
#else
    // tanh(x) = 1 - 2/(e^{2x}+1),  e^{2x} = 2^{x*2*log2(e)}
    float cx = fminf(fmaxf(x, -9.0f), 9.0f);
#if __has_builtin(__builtin_amdgcn_exp2f)
    float e = __builtin_amdgcn_exp2f(cx * 2.8853900817779268f);
#else
    float e = exp2f(cx * 2.8853900817779268f);
#endif
#if __has_builtin(__builtin_amdgcn_rcpf)
    return 1.0f - 2.0f * __builtin_amdgcn_rcpf(e + 1.0f);
#else
    return 1.0f - 2.0f / (e + 1.0f);
#endif
#endif
}

// ---------------- WMMA fragment loaders -----------------------------------
// A fragment (16x32 f16, row-major staging buffer with row stride HID halves):
//   lanes 0-15 : M=lane,     K = kb*32 + {0..15}
//   lanes 16-31: M=lane-16,  K = kb*32 + {16..31}
__device__ __forceinline__ v16h load_afrag(const _Float16* base, int kb, int lm, int sel) {
    const _Float16* p = base + lm * HID + kb * 32 + sel * 16;
    union { v16h v; v8h h[2]; } u;
    u.h[0] = *(const v8h*)(p);
    u.h[1] = *(const v8h*)(p + 8);
    return u.v;
}

// B fragment (32x16 f16) from a TRANSPOSED weight buffer Wt[n][k] (stride SK):
//   lane l: N = n*16 + (l%16), halves K = kb*32 + (l/16)*16 + 0..15  (contiguous)
template<int KB, int SK>
__device__ __forceinline__ void gemm_acc(const _Float16* A, const _Float16* Wt,
                                         v8f acc[8], int lm, int sel) {
#pragma unroll
    for (int kb = 0; kb < KB; ++kb) {
        v16h a = load_afrag(A, kb, lm, sel);
#pragma unroll
        for (int n = 0; n < 8; ++n) {
            v16h b = *(const v16h*)(Wt + (n * 16 + lm) * SK + kb * 32 + sel * 16);
            acc[n] = __builtin_amdgcn_wmma_f32_16x16x32_f16(
                false, a, false, b, (short)0, acc[n], false, false);
        }
    }
}

__device__ __forceinline__ void bcast_init(v8f acc[8], const float bv[8]) {
#pragma unroll
    for (int n = 0; n < 8; ++n) {
        v8f v;
#pragma unroll
        for (int r = 0; r < 8; ++r) v[r] = bv[n];
        acc[n] = v;
    }
}

#if USE_TDM
// ---------------- Tensor Data Mover: 2D tile dW[step] -> LDS ---------------
// D# per CDNA5 ISA:
//  group0: [1:0]=count=1, [63:32]=lds_addr, [120:64]=global_addr, [127:126]=type=2
//  group1: [17:16]=data_size(2 -> 4B), [79:48]=tensor_dim0=128,
//          [111:80]=tensor_dim1=256, [127:112]=tile_dim0=128,
//          [143:128]=tile_dim1=16, [207:160]=tensor_dim0_stride=128
//  remaining groups: zero (tile_dim2..4 unused -> 2D tensor)
__device__ __forceinline__ void tdm_load_dw_tile(unsigned long long gaddr, unsigned lds_off) {
    v4u g0;
    g0.x = 1u;                                        // count=1, user descriptor
    g0.y = lds_off;                                   // LDS byte offset
    g0.z = (unsigned)gaddr;                           // global_addr[31:0]
    g0.w = (unsigned)(gaddr >> 32) | (2u << 30);      // global_addr[56:32] | type=2
    v8i g1;
    g1[0] = (int)(2u << 16);                          // data_size = 4 bytes
    g1[1] = (int)(128u << 16);                        // tensor_dim0 = 128 (lo 16)
    g1[2] = (int)(256u << 16);                        // dim0 hi=0 | tensor_dim1 = 256 (lo 16)
    g1[3] = (int)(128u << 16);                        // dim1 hi=0 | tile_dim0 = 128
    g1[4] = 16;                                       // tile_dim1 = 16
    g1[5] = 128;                                      // tensor_dim0_stride = 128 (lo 32)
    g1[6] = 0;
    g1[7] = 0;
    v4i z4 = {0, 0, 0, 0};
    v8i z8 = {0, 0, 0, 0, 0, 0, 0, 0};
    // 6-arg form (clang-23 / therock-10.0 headers)
    __builtin_amdgcn_tensor_load_to_lds(g0, g1, z4, z4, z8, 0);
}
#endif

// ---------------- kernel 1: the full EM scan -------------------------------
// grid = 16 blocks x 32 threads (one wave per block, 16 trajectories each).
// Trajectories are row-independent -> no synchronization anywhere; DS ops
// within a wave are in-order, so the LDS staging needs no barriers.
__global__ __launch_bounds__(32, 1) void sde_scan_kernel(
    const float* __restrict__ times, const float* __restrict__ x0,
    const float* __restrict__ dW,    const int*   __restrict__ findex,
    const float* __restrict__ Winit, const float* __restrict__ binit,
    const float* __restrict__ Wf1,   const float* __restrict__ bf1,
    const float* __restrict__ Wf2,   const float* __restrict__ bf2,
    const float* __restrict__ Wg1,   const float* __restrict__ bg1,
    const float* __restrict__ Wg2,   const float* __restrict__ bg2,
    float* __restrict__ zf_out)
{
    __shared__ __align__(32) _Float16 s_wf1[HID * HID];   // W^T [n][k], f16
    __shared__ __align__(32) _Float16 s_wf2[HID * HID];
    __shared__ __align__(32) _Float16 s_wg1[HID * HID];
    __shared__ __align__(32) _Float16 s_wg2[HID * HID];
    __shared__ __align__(32) _Float16 s_wi [HID * IN_CH]; // Winit^T [n][k]
    __shared__ __align__(32) _Float16 s_z  [16 * HID];    // A staging (z)
    __shared__ __align__(32) _Float16 s_h  [16 * HID];    // A staging (hidden)
#if USE_TDM
    __shared__ __align__(16) float    s_dw [2][16 * HID]; // TDM double buffer (8KB each)
#endif

    const int lane = threadIdx.x;
    const int lm   = lane & 15;
    const int sel  = lane >> 4;
    const int b0   = blockIdx.x * 16;

    // ---- one-time: transpose weights into LDS as f16 ----
    for (int i = lane; i < HID * HID; i += 32) {
        int k = i >> 7, n = i & 127;
        s_wf1[n * HID + k] = (_Float16)Wf1[i];
        s_wf2[n * HID + k] = (_Float16)Wf2[i];
        s_wg1[n * HID + k] = (_Float16)Wg1[i];
        s_wg2[n * HID + k] = (_Float16)Wg2[i];
    }
    for (int i = lane; i < IN_CH * HID; i += 32) {
        int k = i >> 7, n = i & 127;
        s_wi[n * IN_CH + k] = (_Float16)Winit[i];
    }
    // ---- stage this block's x0 tile (16 x 32) into s_z rows ----
    for (int i = lane; i < 16 * IN_CH; i += 32) {
        int r = i >> 5, c = i & 31;
        s_z[r * HID + c] = (_Float16)x0[(b0 + r) * IN_CH + c];
    }

    // ---- dt = max(min(diff(times)), 0.001), sdt = sqrt(dt) ----
    float dmin = 1e30f;
    for (int i = lane; i < T_STEPS - 1; i += 32)
        dmin = fminf(dmin, times[i + 1] - times[i]);
#pragma unroll
    for (int off = 16; off > 0; off >>= 1)
        dmin = fminf(dmin, __shfl_xor(dmin, off, 32));
    const float dt  = fmaxf(dmin, 0.001f);
    const float sdt = sqrtf(dt);

    // ---- per-lane bias vectors (C-layout: value depends only on N = n*16+lm)
    float bf1v[8], bf2v[8], bg1v[8], bg2v[8], biv[8];
    int   fidx[8];
#pragma unroll
    for (int n = 0; n < 8; ++n) {
        bf1v[n] = bf1[n * 16 + lm];  bf2v[n] = bf2[n * 16 + lm];
        bg1v[n] = bg1[n * 16 + lm];  bg2v[n] = bg2[n * 16 + lm];
        biv [n] = binit[n * 16 + lm];
    }
#pragma unroll
    for (int r = 0; r < 8; ++r) fidx[r] = findex[b0 + r + 8 * sel];

    // ---- z0 = x0 @ Winit + binit   (single K-block GEMM, K=32) ----
    v8f z[8], zf[8];
    bcast_init(z, biv);
    gemm_acc<1, IN_CH>(s_z, s_wi, z, lm, sel);
#pragma unroll
    for (int n = 0; n < 8; ++n) zf[n] = z[n];   // covers final_index == 0

#if USE_TDM
    // ---- prologue: DMA dW tile for step t=1 into buffer 0 ----
    unsigned long long ga = (unsigned long long)(size_t)(dW + (size_t)b0 * HID);
    const unsigned lds_dw[2] = { (unsigned)(size_t)(void*)&s_dw[0][0],
                                 (unsigned)(size_t)(void*)&s_dw[1][0] };
    tdm_load_dw_tile(ga, lds_dw[0]);
#endif

    // ---- Euler-Maruyama scan: t = 1 .. T-1, dw = dW[t-1] ----
#pragma unroll 1
    for (int t = 1; t < T_STEPS; ++t) {
#if !USE_TDM
        // fallback: stream this step's dW tile with per-lane NT loads
        float dwv[8][8];
        const float* dwp = dW + (size_t)(t - 1) * (BATCH * HID) + (size_t)b0 * HID;
#pragma unroll
        for (int n = 0; n < 8; ++n)
#pragma unroll
            for (int r = 0; r < 8; ++r)
                dwv[n][r] = __builtin_nontemporal_load(
                    dwp + (r + 8 * sel) * HID + n * 16 + lm);
#endif

        // z (f32, C-layout regs) -> s_z (f16, row-major) for A fragments
#pragma unroll
        for (int n = 0; n < 8; ++n)
#pragma unroll
            for (int r = 0; r < 8; ++r)
                s_z[(r + 8 * sel) * HID + n * 16 + lm] = (_Float16)z[n][r];

        v8f acc[8];

        // f path: hf = tanh(z@Wf1 + bf1);  f = hf@Wf2 + bf2
        bcast_init(acc, bf1v);
        gemm_acc<4, HID>(s_z, s_wf1, acc, lm, sel);
#pragma unroll
        for (int n = 0; n < 8; ++n)
#pragma unroll
            for (int r = 0; r < 8; ++r)
                s_h[(r + 8 * sel) * HID + n * 16 + lm] = (_Float16)fast_tanh(acc[n][r]);

        bcast_init(acc, bf2v);
        gemm_acc<4, HID>(s_h, s_wf2, acc, lm, sel);
#pragma unroll
        for (int n = 0; n < 8; ++n)
#pragma unroll
            for (int r = 0; r < 8; ++r)
                z[n][r] = fmaf(dt, acc[n][r], z[n][r]);   // z += f*dt (g uses s_z copy)

#if USE_TDM
        // TDM for this step was issued a full step ago -> wait is cheap here,
        // then pull the tile out of LDS in C layout.
        __builtin_amdgcn_s_wait_tensorcnt(0);
        float dwv[8][8];
        {
            const float* dwbuf = &s_dw[(t - 1) & 1][0];
#pragma unroll
            for (int n = 0; n < 8; ++n)
#pragma unroll
                for (int r = 0; r < 8; ++r)
                    dwv[n][r] = dwbuf[(r + 8 * sel) * HID + n * 16 + lm];
        }
#endif

        // g path: g = tanh(tanh(z@Wg1+bg1)@Wg2 + bg2)   (reads pre-update z from s_z)
        bcast_init(acc, bg1v);
        gemm_acc<4, HID>(s_z, s_wg1, acc, lm, sel);
#pragma unroll
        for (int n = 0; n < 8; ++n)
#pragma unroll
            for (int r = 0; r < 8; ++r)
                s_h[(r + 8 * sel) * HID + n * 16 + lm] = (_Float16)fast_tanh(acc[n][r]);

        bcast_init(acc, bg2v);
        gemm_acc<4, HID>(s_h, s_wg2, acc, lm, sel);

        // z += g * (sdt * dw);  then capture rows whose final_index == t
#pragma unroll
        for (int n = 0; n < 8; ++n)
#pragma unroll
            for (int r = 0; r < 8; ++r)
                z[n][r] = fmaf(fast_tanh(acc[n][r]), sdt * dwv[n][r], z[n][r]);

#pragma unroll
        for (int r = 0; r < 8; ++r) {
            bool cap = (fidx[r] == t);
#pragma unroll
            for (int n = 0; n < 8; ++n)
                zf[n][r] = cap ? z[n][r] : zf[n][r];
        }

#if USE_TDM
        // issue the DMA for step t+1 into the other buffer (double-buffered;
        // drain this wave's DS reads of the target buffer region first)
        if (t + 1 < T_STEPS) {
            asm volatile("s_wait_dscnt 0x0" ::: "memory");
            ga += (unsigned long long)BATCH * HID * 4u;
            tdm_load_dw_tile(ga, lds_dw[t & 1]);
        }
#endif
    }

    // ---- write gathered states zf [16 x 128] to workspace (row-major) ----
#pragma unroll
    for (int n = 0; n < 8; ++n)
#pragma unroll
        for (int r = 0; r < 8; ++r)
            zf_out[(size_t)(b0 + r + 8 * sel) * HID + n * 16 + lm] = zf[n][r];
}

// ---------------- kernel 2: readout Linear -> BN -> ReLU -> Linear ---------
__global__ __launch_bounds__(256) void readout_kernel(
    const float* __restrict__ zf,
    const float* __restrict__ W1, const float* __restrict__ b1,
    const float* __restrict__ gamma, const float* __restrict__ beta,
    const float* __restrict__ W2, const float* __restrict__ b2,
    float* __restrict__ out)
{
    __shared__ float sW1[HID * HID];     // 64 KB
    __shared__ float hbuf[BATCH * HID];  // 128 KB (for cross-batch BN stats)
    __shared__ float smean[HID], sinv[HID];

    const int tid = threadIdx.x;   // = batch row b
    for (int i = tid; i < HID * HID; i += 256) sW1[i] = W1[i];
    __syncthreads();

    // h = zf[b] @ W1 + b1  (k-outer; sW1 row reads broadcast across the wave)
    float hacc[HID];
#pragma unroll
    for (int j = 0; j < HID; ++j) hacc[j] = b1[j];
#pragma unroll 1
    for (int k = 0; k < HID; ++k) {
        float zv = zf[tid * HID + k];
#pragma unroll
        for (int j = 0; j < HID; ++j) hacc[j] = fmaf(zv, sW1[k * HID + j], hacc[j]);
    }
#pragma unroll
    for (int j = 0; j < HID; ++j) hbuf[tid * HID + j] = hacc[j];
    __syncthreads();

    // batch statistics per feature
    if (tid < HID) {
        float s = 0.0f, sq = 0.0f;
        for (int b = 0; b < BATCH; ++b) {
            float v = hbuf[b * HID + tid];
            s += v; sq = fmaf(v, v, sq);
        }
        float mean = s * (1.0f / BATCH);
        float var  = sq * (1.0f / BATCH) - mean * mean;
        smean[tid] = mean;
        sinv[tid]  = rsqrtf(var + BN_EPS);
    }
    __syncthreads();

    // out = relu(gamma*(h-mean)*inv + beta) @ W2 + b2
    float o[OUT_CH];
#pragma unroll
    for (int c = 0; c < OUT_CH; ++c) o[c] = b2[c];
#pragma unroll
    for (int j = 0; j < HID; ++j) {
        float hn = fmaf(gamma[j] * (hacc[j] - smean[j]), sinv[j], beta[j]);
        hn = fmaxf(hn, 0.0f);
#pragma unroll
        for (int c = 0; c < OUT_CH; ++c) o[c] = fmaf(hn, W2[j * OUT_CH + c], o[c]);
    }
#pragma unroll
    for (int c = 0; c < OUT_CH; ++c) out[tid * OUT_CH + c] = o[c];
}

// ---------------- launcher --------------------------------------------------
extern "C" void kernel_launch(void* const* d_in, const int* in_sizes, int n_in,
                              void* d_out, int out_size, void* d_ws, size_t ws_size,
                              hipStream_t stream) {
    const float* times  = (const float*)d_in[0];
    const float* x0     = (const float*)d_in[1];
    const float* dW     = (const float*)d_in[2];
    const int*   findex = (const int*)  d_in[3];
    const float* Winit  = (const float*)d_in[4];
    const float* binit  = (const float*)d_in[5];
    const float* Wf1    = (const float*)d_in[6];
    const float* bf1    = (const float*)d_in[7];
    const float* Wf2    = (const float*)d_in[8];
    const float* bf2    = (const float*)d_in[9];
    const float* Wg1    = (const float*)d_in[10];
    const float* bg1    = (const float*)d_in[11];
    const float* Wg2    = (const float*)d_in[12];
    const float* bg2    = (const float*)d_in[13];
    const float* W1     = (const float*)d_in[14];
    const float* b1     = (const float*)d_in[15];
    const float* gamma  = (const float*)d_in[16];
    const float* beta   = (const float*)d_in[17];
    const float* W2     = (const float*)d_in[18];
    const float* b2     = (const float*)d_in[19];

    float* zf = (float*)d_ws;   // BATCH*HID floats = 128 KB scratch

    sde_scan_kernel<<<dim3(BATCH / 16), dim3(32), 0, stream>>>(
        times, x0, dW, findex, Winit, binit,
        Wf1, bf1, Wf2, bf2, Wg1, bg1, Wg2, bg2, zf);

    readout_kernel<<<dim3(1), dim3(256), 0, stream>>>(
        zf, W1, b1, gamma, beta, W2, b2, (float*)d_out);
}
